// CrossAttn_13950053778233
// MI455X (gfx1250) — compile-verified, matrix operand order checked by reading, not run on hardware
//
#include <hip/hip_runtime.h>
#include <hip/hip_bf16.h>

// ---------------------------------------------------------------------------
// CrossAttn on MI455X (gfx1250): all 5 GEMMs via v_wmma_f32_16x16x32_bf16,
// fed by GLOBAL_LOAD_ASYNC_TO_LDS_B128 (ASYNCcnt) with double-buffered LDS.
//   q  = y @ Wq^T + bq            [B*Nq, CH]
//   k  = x @ Wk^T + bk            [B*Nk, CH]
//   vT = Wv @ x^T + bv (per-row)  [B][C, Nk]
//   dots = q @ k^T * scale        [B][Nq, Nk]  f32
//   attn = softmax over QUERY axis (per key column m)  -> bf16
//   out  = attn @ (vT)^T          -> f32 d_out
// ---------------------------------------------------------------------------

typedef __attribute__((ext_vector_type(16))) __bf16 v16bf;
typedef __attribute__((ext_vector_type(8)))  __bf16 v8bf;
typedef __attribute__((ext_vector_type(8)))  float  v8f;

// Typed pointers for the async-copy builtin: param is pointer-to-int4
// (16B = b128) in the appropriate address space.
typedef int vint4 __attribute__((vector_size(16)));
typedef __attribute__((address_space(1))) vint4 g_vint4;   // global
typedef __attribute__((address_space(3))) vint4 l_vint4;   // LDS

#define BM 128
#define BN 128
#define BK 32
#define LDT 40   // padded LDS row stride in bf16 elements (80B, 16B-aligned)

__device__ __forceinline__ unsigned short f2bf(float x) {
  unsigned int u = __float_as_uint(x);
  u += 0x7FFFu + ((u >> 16) & 1u);   // round-to-nearest-even
  return (unsigned short)(u >> 16);
}

union FragU { v16bf v; v8bf h[2]; };

// Fragment load per ISA 16-bit A/B layout:
// lane r=lane&15 owns row r; halfk=lane>>4 selects K 0-7/16-23 vs 8-15/24-31.
__device__ __forceinline__ v16bf ld_frag(const unsigned short* lds, int row, int halfk) {
  const __bf16* p = (const __bf16*)(lds + row * LDT) + halfk * 8;
  FragU f;
  f.h[0] = *(const v8bf*)(p);        // K = halfk*8 .. +7
  f.h[1] = *(const v8bf*)(p + 16);   // K = 16 + halfk*8 .. +7
  return f.v;
}

// 16B global -> LDS async copy (gfx1250, tracked by ASYNCcnt)
__device__ __forceinline__ void async_cp16(const unsigned short* g, unsigned short* l) {
  __builtin_amdgcn_global_load_async_to_lds_b128((g_vint4*)g, (l_vint4*)l, 0, 0);
}

// Issue async copies for one (A,B) K-tile: 4 ops per thread (2 A + 2 B).
__device__ __forceinline__ void stage_tile(const unsigned short* gA, long lda,
                                           const unsigned short* gB, long ldb,
                                           unsigned short* lA, unsigned short* lB,
                                           int k0, int tid) {
#pragma unroll
  for (int rr = 0; rr < 2; rr++) {
    int id = tid + rr * 256, row = id >> 2, cc = (id & 3) * 8;
    async_cp16(gA + (long)row * lda + k0 + cc, lA + row * LDT + cc);
  }
#pragma unroll
  for (int rr = 0; rr < 2; rr++) {
    int id = tid + rr * 256, row = id >> 2, cc = (id & 3) * 8;
    async_cp16(gB + (long)row * ldb + k0 + cc, lB + row * LDT + cc);
  }
}

// One K-tile of compute: load 4 A-frags + 2 B-frags from LDS, 8 WMMAs.
__device__ __forceinline__ void compute_tile(const unsigned short* lA,
                                             const unsigned short* lB,
                                             int wm, int wn, int r, int hk,
                                             v8f (&acc)[4][2]) {
  v16bf af[4], bfr[2];
#pragma unroll
  for (int i = 0; i < 4; i++) af[i]  = ld_frag(lA, wm * 64 + i * 16 + r, hk);
#pragma unroll
  for (int j = 0; j < 2; j++) bfr[j] = ld_frag(lB, wn * 32 + j * 16 + r, hk);
#pragma unroll
  for (int i = 0; i < 4; i++)
#pragma unroll
    for (int j = 0; j < 2; j++)
      acc[i][j] = __builtin_amdgcn_wmma_f32_16x16x32_bf16(
          false, af[i], false, bfr[j], (short)0, acc[i][j], false, false);
}

// ---------------------------------------------------------------------------
// f32 -> bf16 conversion (vectorized x4)
// ---------------------------------------------------------------------------
__global__ void cvt_f32_bf16_v4(const float4* __restrict__ in,
                                ushort4* __restrict__ out, long n4) {
  long i = (long)blockIdx.x * blockDim.x + threadIdx.x;
  long stride = (long)gridDim.x * blockDim.x;
  for (; i < n4; i += stride) {
    float4 f = in[i];
    out[i] = make_ushort4(f2bf(f.x), f2bf(f.y), f2bf(f.z), f2bf(f.w));
  }
}

// ---------------------------------------------------------------------------
// Tiled WMMA GEMM:  C[M,N] = A[M,K] * B[N,K]^T  (A,B bf16 row-major)
// BIAS_MODE: 0 none, 1 per-N, 2 per-M.  OUT_BF16: store bf16 else f32.
// grid = (N/BN, M/BM, batch); sA/sB/sC per-batch element strides.
// Block 256 threads = 8 wave32s in 2(M)x4(N); wave tile 64x32 = 4x2 frags.
// Steady-state loop is branch-free: stage(next) -> wait(4) -> barrier ->
// compute(cur) -> barrier; the last tile is peeled (wait 0).
// ---------------------------------------------------------------------------
template <int BIAS_MODE, bool OUT_BF16>
__global__ __launch_bounds__(256)
void gemm_bf16_tn(const unsigned short* __restrict__ A, long lda, long sA,
                  const unsigned short* __restrict__ B, long ldb, long sB,
                  void* __restrict__ Cv, long ldc, long sC,
                  const float* __restrict__ bias, float scale, int K) {
  __shared__ __align__(16) unsigned short sAb[2][BM * LDT];
  __shared__ __align__(16) unsigned short sBb[2][BN * LDT];

  const int tid  = threadIdx.x;
  const int lane = tid & 31;
  const int wave = tid >> 5;
  const int wm   = wave & 1;       // 2 wave-rows -> 64 M each
  const int wn   = wave >> 1;      // 4 wave-cols -> 32 N each
  const int r    = lane & 15;
  const int hk   = lane >> 4;

  const long m0 = (long)blockIdx.y * BM;
  const long n0 = (long)blockIdx.x * BN;
  A += (long)blockIdx.z * sA + m0 * lda;
  B += (long)blockIdx.z * sB + n0 * ldb;

  v8f acc[4][2];
#pragma unroll
  for (int i = 0; i < 4; i++)
#pragma unroll
    for (int j = 0; j < 2; j++) acc[i][j] = v8f{};

  // Prologue: async-stage tile 0 into buffer 0.
  stage_tile(A, lda, B, ldb, sAb[0], sBb[0], 0, tid);

  // Steady state: always has a next tile (last tile peeled below).
  int buf = 0;
  int k0  = 0;
  for (; k0 + BK < K; k0 += BK, buf ^= 1) {
    stage_tile(A, lda, B, ldb, sAb[buf ^ 1], sBb[buf ^ 1], k0 + BK, tid);
    asm volatile("s_wait_asynccnt 4" ::: "memory");  // current tile's 4 ops done
    __syncthreads();
    compute_tile(sAb[buf], sBb[buf], wm, wn, r, hk, acc);
    __syncthreads();  // everyone done reading buf before it is re-staged
  }
  // Final tile.
  asm volatile("s_wait_asynccnt 0" ::: "memory");
  __syncthreads();
  compute_tile(sAb[buf], sBb[buf], wm, wn, r, hk, acc);

  // Epilogue: C/D layout = VGPR e, lane group hk -> m = base + e, n = r
  const long zc = (long)blockIdx.z * sC;
  float* Cf          = (float*)Cv;
  unsigned short* Cb = (unsigned short*)Cv;
#pragma unroll
  for (int i = 0; i < 4; i++) {
#pragma unroll
    for (int j = 0; j < 2; j++) {
      const long mbase = m0 + wm * 64 + i * 16 + 8 * hk;
      const long n     = n0 + wn * 32 + j * 16 + r;
      const float bn   = (BIAS_MODE == 1) ? bias[n] : 0.0f;
#pragma unroll
      for (int e = 0; e < 8; e++) {
        const long m = mbase + e;
        float v = acc[i][j][e] * scale + bn;
        if (BIAS_MODE == 2) v += bias[m];
        const long idx = zc + m * ldc + n;
        if (OUT_BF16) Cb[idx] = f2bf(v);
        else          Cf[idx] = v;
      }
    }
  }
}

// ---------------------------------------------------------------------------
// Column softmax (over query axis n) per (batch, key m).
// Block: 256 threads = 64 columns x 4 row-segments; LDS combine of partial
// online max/sum, then each segment normalizes + stores bf16.
// ---------------------------------------------------------------------------
__global__ __launch_bounds__(256)
void softmax_cols(const float* __restrict__ dots, unsigned short* __restrict__ attn,
                  int Nq, long Nk) {
  __shared__ float smax[4][64];
  __shared__ float ssum[4][64];
  const int c  = threadIdx.x & 63;
  const int rg = threadIdx.x >> 6;            // 0..3
  const long m = (long)blockIdx.x * 64 + c;
  const float* dp    = dots + (long)blockIdx.y * Nq * Nk + m;
  unsigned short* ap = attn + (long)blockIdx.y * Nq * Nk + m;
  const int seg = Nq >> 2;
  const int nA = rg * seg, nB = nA + seg;

  float mx = -3.4e38f, s = 0.0f;
  for (int n = nA; n < nB; n++) {
    float v = dp[(long)n * Nk];
    if (v > mx) { s *= __expf(mx - v); mx = v; }
    s += __expf(v - mx);
  }
  smax[rg][c] = mx;
  ssum[rg][c] = s;
  __syncthreads();

  float M = smax[0][c];
#pragma unroll
  for (int g = 1; g < 4; g++) M = fmaxf(M, smax[g][c]);
  float S = 0.0f;
#pragma unroll
  for (int g = 0; g < 4; g++) S += ssum[g][c] * __expf(smax[g][c] - M);
  const float inv = 1.0f / S;

  for (int n = nA; n < nB; n++) {
    float v = __expf(dp[(long)n * Nk] - M) * inv;
    ap[(long)n * Nk] = f2bf(v);
  }
}

// ---------------------------------------------------------------------------
extern "C" void kernel_launch(void* const* d_in, const int* in_sizes, int n_in,
                              void* d_out, int out_size, void* d_ws, size_t ws_size,
                              hipStream_t stream) {
  const float* y  = (const float*)d_in[0];
  const float* x  = (const float*)d_in[1];
  const float* Wq = (const float*)d_in[2];
  const float* bq = (const float*)d_in[3];
  const float* Wk = (const float*)d_in[4];
  const float* bk = (const float*)d_in[5];
  const float* Wv = (const float*)d_in[6];
  const float* bv = (const float*)d_in[7];
  float* out = (float*)d_out;

  const long B = 8, Nq = 2048, Nk = 2048, Cc = 1024, CH = 512;
  const float scale = 0.044194173824159216f;  // (C/2)^-0.5

  char* ws = (char*)d_ws;
  size_t off = 0;
  auto wsalloc = [&](size_t bytes) -> void* {
    void* p = ws + off;
    off += (bytes + 255) & ~(size_t)255;
    return p;
  };
  unsigned short* yh   = (unsigned short*)wsalloc(B * Nq * Cc * 2);
  unsigned short* xh   = (unsigned short*)wsalloc(B * Nk * Cc * 2);
  unsigned short* wqh  = (unsigned short*)wsalloc(CH * Cc * 2);
  unsigned short* wkh  = (unsigned short*)wsalloc(CH * Cc * 2);
  unsigned short* wvh  = (unsigned short*)wsalloc(Cc * Cc * 2);
  unsigned short* qh   = (unsigned short*)wsalloc(B * Nq * CH * 2);
  unsigned short* kh   = (unsigned short*)wsalloc(B * Nk * CH * 2);
  unsigned short* vth  = (unsigned short*)wsalloc(B * Cc * Nk * 2);
  float*          dots = (float*)wsalloc(B * Nq * Nk * 4);
  unsigned short* ath  = (unsigned short*)wsalloc(B * Nq * Nk * 2);

  dim3 blk(256);

  // f32 -> bf16 conversions
  cvt_f32_bf16_v4<<<2048, blk, 0, stream>>>((const float4*)y,  (ushort4*)yh,  B * Nq * Cc / 4);
  cvt_f32_bf16_v4<<<2048, blk, 0, stream>>>((const float4*)x,  (ushort4*)xh,  B * Nk * Cc / 4);
  cvt_f32_bf16_v4<<<256,  blk, 0, stream>>>((const float4*)Wq, (ushort4*)wqh, CH * Cc / 4);
  cvt_f32_bf16_v4<<<256,  blk, 0, stream>>>((const float4*)Wk, (ushort4*)wkh, CH * Cc / 4);
  cvt_f32_bf16_v4<<<256,  blk, 0, stream>>>((const float4*)Wv, (ushort4*)wvh, Cc * Cc / 4);

  // q = y @ Wq^T + bq   -> bf16 [B*Nq, CH]
  gemm_bf16_tn<1, true><<<dim3(CH / BN, (B * Nq) / BM, 1), blk, 0, stream>>>(
      yh, Cc, 0, wqh, Cc, 0, qh, CH, 0, bq, 1.0f, (int)Cc);
  // k = x @ Wk^T + bk   -> bf16 [B*Nk, CH]
  gemm_bf16_tn<1, true><<<dim3(CH / BN, (B * Nk) / BM, 1), blk, 0, stream>>>(
      xh, Cc, 0, wkh, Cc, 0, kh, CH, 0, bk, 1.0f, (int)Cc);
  // vT[b] = Wv @ x[b]^T + bv(per-M)  -> bf16 [B][Cc, Nk]
  gemm_bf16_tn<2, true><<<dim3(Nk / BN, Cc / BM, B), blk, 0, stream>>>(
      wvh, Cc, 0, xh, Cc, Nk * Cc, vth, Nk, Cc * Nk, bv, 1.0f, (int)Cc);
  // dots[b] = q[b] @ k[b]^T * scale  -> f32 [B][Nq, Nk]
  gemm_bf16_tn<0, false><<<dim3(Nk / BN, Nq / BM, B), blk, 0, stream>>>(
      qh, CH, Nq * CH, kh, CH, Nk * CH, dots, Nk, Nq * Nk, nullptr, scale, (int)CH);
  // attn = softmax over query axis -> bf16
  softmax_cols<<<dim3(Nk / 64, B), blk, 0, stream>>>(dots, ath, (int)Nq, Nk);
  // out[b] = attn[b] @ vT[b]^T  -> f32 d_out
  gemm_bf16_tn<0, false><<<dim3(Cc / BN, Nq / BM, B), blk, 0, stream>>>(
      ath, Nk, Nq * Nk, vth, Nk, Cc * Nk, out, Cc, Nq * Cc, nullptr, 1.0f, (int)Nk);

  (void)in_sizes; (void)n_in; (void)out_size; (void)ws_size;
}
